// GRU_13168369729783
// MI455X (gfx1250) — matrix operand order repaired and checked by reading, not compiled, not run on hardware
//
#include <hip/hip_runtime.h>
#include <math.h>

// ---------------------------------------------------------------------------
// 2-layer GRU for MI455X (gfx1250).  B=32, S=2048, IN=H=256.
//
//  * Parallel phase: gi = act @ Wih^T + bih as whole-GPU WMMA GEMMs
//    (bf16 in, f32 accumulate).
//  * Sequential phase: one persistent 512-thread workgroup (16 wave32s,
//    4 waves/SIMD => ~256 VGPR/wave budget, no spills) iterates t=0..2047.
//    gh = h @ Whh^T via v_wmma_f32_16x16x32_bf16 (768 WMMA/step).
//    Whh fully resident:
//      - 16 N-tiles (128 KB bf16) in per-wave VGPRs (wave w owns nt=w)
//      - 32 N-tiles (256 KB) in LDS, preloaded once via
//        global_load_async_to_lds_b128 in B-fragment layout
//    => zero global Whh traffic in the time loop.  A fragments loaded once
//    per k-slice and reused by all 3 N-units (6 WMMA per k-slice).
//  * Dynamic LDS: h(16K) + gh(48K) + Whh cache(256K) = 320 KB (full WGP).
// ---------------------------------------------------------------------------

typedef __attribute__((ext_vector_type(16))) __bf16 v16bf;
typedef __attribute__((ext_vector_type(8)))  float  v8f;

#define B_   32
#define S_   2048
#define IN_  256
#define H_   256
#define G3H  768   // 3*H

union FragBF {
    v16bf  v;
    __bf16 e[16];
    uint4  u[2];
};

__device__ __forceinline__ float sigmoidf_(float x) {
    return 1.0f / (1.0f + __expf(-x));
}

__device__ __forceinline__ v8f wmma_bf16(const FragBF& a, const FragBF& b, v8f c) {
    return __builtin_amdgcn_wmma_f32_16x16x32_bf16(
        false, a.v, false, b.v, (short)0, c, false, false);
}

// --------------------------- f32 -> bf16 convert ---------------------------
__global__ void cvt_f32_bf16(const float* __restrict__ in,
                             __bf16* __restrict__ out, int n) {
    int i = blockIdx.x * blockDim.x + threadIdx.x;
    if (i < n) out[i] = (__bf16)in[i];
}

// ---------------------------------------------------------------------------
// gi[t*B+b][0:768] = A[row] @ W^T + bias    (row r -> t = r/B, b = r%B)
//   layer0: A = x (f32, [B,S,IN]),       sb = S*IN, st = IN
//   layer1: A = h0_all (bf16, [S,B,H]),  sb = H,    st = B*H
// ---------------------------------------------------------------------------
template <bool ABF16>
__global__ __launch_bounds__(256) void gi_gemm(
    const void* __restrict__ A, long sb, long st,
    const __bf16* __restrict__ Wbf, const float* __restrict__ bias,
    __bf16* __restrict__ gi)
{
    const int lane = threadIdx.x & 31;
    const int wave = threadIdx.x >> 5;
    const int mt   = wave & 1;
    const int ntl  = wave >> 1;
    const int rbase = blockIdx.x * 32 + mt * 16;
    const int nbase = blockIdx.y * 64 + ntl * 16;

    const int lm    = lane & 15;
    const int aksel = (lane >= 16) ? 16 : 0;
    const int bksel = (lane >= 16) ? 16 : 0;
    const int msel  = (lane >= 16) ? 8 : 0;

    const int  r    = rbase + lm;
    const long arow = (long)(r % B_) * sb + (long)(r / B_) * st;
    const int  ncol = nbase + lm;

    v8f acc = {};
    #pragma unroll
    for (int kb = 0; kb < 8; ++kb) {
        FragBF a, b;
        if constexpr (ABF16) {
            const __bf16* Ab = (const __bf16*)A + arow + kb * 32;
            a.u[0] = *(const uint4*)(Ab + aksel);
            a.u[1] = *(const uint4*)(Ab + 8 + aksel);
        } else {
            const float* Af = (const float*)A + arow + kb * 32;
            #pragma unroll
            for (int i = 0; i < 8; ++i) a.e[i]     = (__bf16)Af[aksel + i];
            #pragma unroll
            for (int i = 0; i < 8; ++i) a.e[8 + i] = (__bf16)Af[8 + aksel + i];
        }
        const __bf16* Wp = Wbf + (long)ncol * 256 + kb * 32 + bksel;
        b.u[0] = ((const uint4*)Wp)[0];
        b.u[1] = ((const uint4*)Wp)[1];
        acc = wmma_bf16(a, b, acc);
    }

    const float bn = bias[ncol];
    #pragma unroll
    for (int v = 0; v < 8; ++v) {
        const int rr = rbase + v + msel;
        gi[(long)rr * G3H + ncol] = (__bf16)(acc[v] + bn);
    }
}

// ---------------------------------------------------------------------------
// Persistent recurrence kernel.  512 threads = 16 waves, loops t = 0..S-1.
// Wave w owns N-tiles: nt0 = w (B in VGPRs), nt1 = 16+w, nt2 = 32+w (B in LDS)
// ---------------------------------------------------------------------------
__global__ __launch_bounds__(512) void gru_recur(
    const __bf16* __restrict__ gi, const __bf16* __restrict__ Whh,
    const float* __restrict__ bhh,
    __bf16* __restrict__ h_all_out, float* __restrict__ seq_out,
    float* __restrict__ final_out, int layer1)
{
    extern __shared__ __align__(16) char smem[];
    __bf16* h_bf = (__bf16*)smem;                 // 16 KB
    __bf16* gh   = (__bf16*)(smem + 16384);       // 48 KB
    char*   wlds = smem + 65536;                  // 256 KB Whh B-fragment cache

    const int tid  = threadIdx.x;
    const int lane = tid & 31;
    const int wave = tid >> 5;                    // 0..15

    const int lm    = lane & 15;
    const int aksel = (lane >= 16) ? 16 : 0;
    const int bksel = (lane >= 16) ? 16 : 0;
    const int msel  = (lane >= 16) ? 8 : 0;

    // ---- one-time: zero h ----
    for (int i = tid; i < B_ * H_; i += 512) h_bf[i] = (__bf16)0.0f;

    // ---- one-time: preload B fragments for nt0 = wave into registers ----
    FragBF Breg[8];
    {
        const __bf16* wp0 = Whh + (long)(wave * 16 + lm) * H_ + bksel;
        #pragma unroll
        for (int kb = 0; kb < 8; ++kb) {
            Breg[kb].u[0] = ((const uint4*)(wp0 + kb * 32))[0];
            Breg[kb].u[1] = ((const uint4*)(wp0 + kb * 32))[1];
        }
    }

    // ---- one-time: async-preload nt = 16..47 B fragments into LDS ----
    // unit u = nt-16 (0..31); fragment i = ((u*8+kb)*32 + lane): 32B each
    for (int i = tid; i < 8192; i += 512) {
        const int u  = i >> 8;
        const int kb = (i >> 5) & 7;
        const int ln = i & 31;
        const int n  = (16 + u) * 16 + (ln & 15);
        const int kg = kb * 32 + ((ln >= 16) ? 16 : 0);
        const __bf16* src = Whh + (long)n * H_ + kg;
        const unsigned dst = (unsigned)(uintptr_t)(wlds + (long)i * 32);
        asm volatile("global_load_async_to_lds_b128 %0, %1, off"
                     :: "v"(dst), "v"(src) : "memory");
        asm volatile("global_load_async_to_lds_b128 %0, %1, off offset:16"
                     :: "v"(dst), "v"(src) : "memory");
    }
    asm volatile("s_wait_asynccnt 0x0" ::: "memory");
    __syncthreads();

    const char* bl1_base = wlds + (long)(wave * 8) * 32 * 32;         // u = wave
    const char* bl2_base = wlds + (long)((16 + wave) * 8) * 32 * 32;  // u = 16+wave

    for (int t = 0; t < S_; ++t) {
        // ================= gh = h @ Whh^T =================
        // 3 N-units x 2 row-tiles; A fragments shared across units per k-slice
        v8f acc00 = {}, acc01 = {};   // nt0 (register B)
        v8f acc10 = {}, acc11 = {};   // nt1 (LDS B)
        v8f acc20 = {}, acc21 = {};   // nt2 (LDS B)
        #pragma unroll
        for (int kb = 0; kb < 8; ++kb) {
            FragBF a0, a1, bl;
            const __bf16* hp0 = &h_bf[(lm)      * H_ + kb * 32];
            const __bf16* hp1 = &h_bf[(16 + lm) * H_ + kb * 32];
            a0.u[0] = *(const uint4*)(hp0 + aksel);
            a0.u[1] = *(const uint4*)(hp0 + 8 + aksel);
            a1.u[0] = *(const uint4*)(hp1 + aksel);
            a1.u[1] = *(const uint4*)(hp1 + 8 + aksel);

            acc00 = wmma_bf16(a0, Breg[kb], acc00);
            acc01 = wmma_bf16(a1, Breg[kb], acc01);

            const char* bp1 = bl1_base + (long)(kb * 32 + lane) * 32;
            bl.u[0] = ((const uint4*)bp1)[0];
            bl.u[1] = ((const uint4*)bp1)[1];
            acc10 = wmma_bf16(a0, bl, acc10);
            acc11 = wmma_bf16(a1, bl, acc11);

            const char* bp2 = bl2_base + (long)(kb * 32 + lane) * 32;
            bl.u[0] = ((const uint4*)bp2)[0];
            bl.u[1] = ((const uint4*)bp2)[1];
            acc20 = wmma_bf16(a0, bl, acc20);
            acc21 = wmma_bf16(a1, bl, acc21);
        }
        {
            const int nc0 = wave * 16 + lm;
            #pragma unroll
            for (int v = 0; v < 8; ++v) {
                gh[(v + msel) * G3H + nc0]            = (__bf16)acc00[v];
                gh[(16 + v + msel) * G3H + nc0]       = (__bf16)acc01[v];
                gh[(v + msel) * G3H + nc0 + 256]      = (__bf16)acc10[v];
                gh[(16 + v + msel) * G3H + nc0 + 256] = (__bf16)acc11[v];
                gh[(v + msel) * G3H + nc0 + 512]      = (__bf16)acc20[v];
                gh[(16 + v + msel) * G3H + nc0 + 512] = (__bf16)acc21[v];
            }
        }

        // prefetch next step's gi rows (48 KB = 384 x 128B lines)
        if (t + 1 < S_ && tid < 384) {
            const char* nx = (const char*)(gi + (long)(t + 1) * B_ * G3H);
            __builtin_prefetch(nx + (long)tid * 128, 0, 0);
        }
        __syncthreads();

        // ================= gating: 8192 elements, 16/thread =================
        const __bf16* git = gi + (long)t * B_ * G3H;
        #pragma unroll
        for (int i = 0; i < 16; ++i) {
            const int e  = tid + i * 512;
            const int bb = e >> 8;
            const int j  = e & 255;
            const long grow = (long)bb * G3H + j;

            const float ir  = (float)git[grow];
            const float iz  = (float)git[grow + 256];
            const float inn = (float)git[grow + 512];
            const float hr  = (float)gh[grow]       + bhh[j];
            const float hz  = (float)gh[grow + 256] + bhh[256 + j];
            const float hn  = (float)gh[grow + 512] + bhh[512 + j];

            const float rg = sigmoidf_(ir + hr);
            const float zg = sigmoidf_(iz + hz);
            const float ng = tanhf(inn + rg * hn);
            const float hprev = (float)h_bf[bb * H_ + j];
            // Faithful to reference: (1-z)*n + h   (h, not z*h)
            const float hnew = (1.0f - zg) * ng + hprev;

            h_bf[bb * H_ + j] = (__bf16)hnew;
            if (layer1)
                seq_out[((long)bb * S_ + t) * H_ + j] = hnew;
            else
                h_all_out[((long)t * B_ + bb) * H_ + j] = (__bf16)hnew;
        }
        __syncthreads();
    }

    for (int i = tid; i < B_ * H_; i += 512)
        final_out[i] = (float)h_bf[i];
}

// ---------------------------------------------------------------------------
extern "C" void kernel_launch(void* const* d_in, const int* in_sizes, int n_in,
                              void* d_out, int out_size, void* d_ws, size_t ws_size,
                              hipStream_t stream) {
    const float* x    = (const float*)d_in[0];
    const float* Wih0 = (const float*)d_in[1];
    const float* Whh0 = (const float*)d_in[2];
    const float* bih0 = (const float*)d_in[3];
    const float* bhh0 = (const float*)d_in[4];
    const float* Wih1 = (const float*)d_in[5];
    const float* Whh1 = (const float*)d_in[6];
    const float* bih1 = (const float*)d_in[7];
    const float* bhh1 = (const float*)d_in[8];

    float* out  = (float*)d_out;
    float* fin0 = out + (long)B_ * S_ * H_;          // final hidden layer 0
    float* fin1 = fin0 + B_ * H_;                    // final hidden layer 1

    // Workspace (bf16): 4 weight matrices + gi buffer (reused) + h0_all.
    const long WSZ = (long)G3H * H_;                 // 196608 per matrix
    __bf16* Wih0b = (__bf16*)d_ws;
    __bf16* Whh0b = Wih0b + WSZ;
    __bf16* Wih1b = Whh0b + WSZ;
    __bf16* Whh1b = Wih1b + WSZ;
    __bf16* gibuf = Whh1b + WSZ;                     // 65536*768 bf16 = 96 MB
    __bf16* h0all = gibuf + (long)B_ * S_ * G3H;     // 65536*256 bf16 = 32 MB

    cvt_f32_bf16<<<768, 256, 0, stream>>>(Wih0, Wih0b, (int)WSZ);
    cvt_f32_bf16<<<768, 256, 0, stream>>>(Whh0, Whh0b, (int)WSZ);
    cvt_f32_bf16<<<768, 256, 0, stream>>>(Wih1, Wih1b, (int)WSZ);
    cvt_f32_bf16<<<768, 256, 0, stream>>>(Whh1, Whh1b, (int)WSZ);

    const dim3 ggrid((B_ * S_) / 32, G3H / 64);      // (2048, 12)
    const size_t lds_bytes = 327680;                 // full 320 KB / WGP

    gi_gemm<false><<<ggrid, 256, 0, stream>>>(
        x, (long)S_ * IN_, (long)IN_, Wih0b, bih0, gibuf);

    gru_recur<<<1, 512, lds_bytes, stream>>>(
        gibuf, Whh0b, bhh0, h0all, nullptr, fin0, /*layer1=*/0);

    gi_gemm<true><<<ggrid, 256, 0, stream>>>(
        h0all, (long)H_, (long)B_ * H_, Wih1b, bih1, gibuf);

    gru_recur<<<1, 512, lds_bytes, stream>>>(
        gibuf, Whh1b, bhh1, nullptr, out, fin1, /*layer1=*/1);
}
